// APPNP_Net_72164040507403
// MI455X (gfx1250) — matrix operand order repaired and checked by reading, not compiled, not run on hardware
//
#include <hip/hip_runtime.h>
#include <hip/hip_bf16.h>

typedef __attribute__((ext_vector_type(16))) __bf16 v16bf;
typedef __attribute__((ext_vector_type(8)))  __bf16 v8bf;
typedef __attribute__((ext_vector_type(8)))  float  v8f;
typedef __attribute__((ext_vector_type(4)))  float  v4f;

#define N_NODES 100000
#define N_EDGES 3200000
#define K_STEPS 10
#define ALPHA   0.1f
#define F_IN    512
#define F_HID   256
#define F_OUT   64

// ---------------------------------------------------------------- weight prep
__global__ __launch_bounds__(256) void prep_w1(const float* __restrict__ W1,
                                               __bf16* __restrict__ w1t) {
    int idx = blockIdx.x * 256 + threadIdx.x;            // 256*512 entries
    if (idx < F_HID * F_IN) {
        int n = idx >> 9;                                // /512
        int k = idx & (F_IN - 1);
        w1t[idx] = (__bf16)W1[(size_t)k * F_HID + n];    // transpose to [N][K]
    }
}

__global__ __launch_bounds__(256) void prep_w2(const float* __restrict__ W2,
                                               __bf16* __restrict__ w2t) {
    int idx = blockIdx.x * 256 + threadIdx.x;            // 64*256 entries
    if (idx < F_OUT * F_HID) {
        int n = idx >> 8;                                // /256
        int k = idx & (F_HID - 1);
        w2t[idx] = (__bf16)W2[(size_t)k * F_OUT + n];
    }
}

// ---------------------------------------------------------------- gcn_norm
__global__ __launch_bounds__(256) void deg_init(float* __restrict__ deg) {
    int i = blockIdx.x * 256 + threadIdx.x;
    if (i < N_NODES) deg[i] = 1.0f;                      // self-loop
}

__global__ __launch_bounds__(256) void deg_scatter(const int* __restrict__ col,
                                                   float* __restrict__ deg) {
    int e = blockIdx.x * 256 + threadIdx.x;
    if (e < N_EDGES) unsafeAtomicAdd(&deg[col[e]], 1.0f);
}

__global__ __launch_bounds__(256) void dinv_kernel(float* __restrict__ d) {
    int i = blockIdx.x * 256 + threadIdx.x;
    if (i < N_NODES) {
        float v = d[i];
        d[i] = (v > 0.0f) ? rsqrtf(v) : 0.0f;
    }
}

__global__ __launch_bounds__(256) void wnorm_kernel(const int* __restrict__ row,
                                                    const int* __restrict__ col,
                                                    const float* __restrict__ dinv,
                                                    float* __restrict__ wn) {
    int e = blockIdx.x * 256 + threadIdx.x;
    if (e < N_EDGES)
        wn[e] = (1.0f - ALPHA) * dinv[row[e]] * dinv[col[e]];
}

// ---------------------------------------------------------------- fused MLP (WMMA)
// One workgroup (8 wave32) computes 16 rows: hidden[16x256] in LDS, then out[16x64].
__global__ __launch_bounds__(256) void mlp_kernel(const float* __restrict__ x,
                                                  const __bf16* __restrict__ w1t,
                                                  const float* __restrict__ b1,
                                                  const __bf16* __restrict__ w2t,
                                                  const float* __restrict__ b2,
                                                  float* __restrict__ h) {
    __shared__ __bf16 hid[16][F_HID + 8];                // pad stride: 264 bf16 = 528B

    const int lane = threadIdx.x & 31;
    const int wave = threadIdx.x >> 5;
    const int m0   = blockIdx.x * 16;

    const int rA  = lane & 15;            // A-matrix row for this lane
    const int kbA = (lane >> 4) * 8;      // A K-offset (ISA 16-bit A 16x32 layout)
    const int cB  = lane & 15;            // B-matrix col for this lane
    const int kbB = (lane >> 4) * 8;      // B K-offset (ISA 16-bit layout, same as A)
    const int mo  = (lane >> 4) * 8;      // C/D row offset for this lane-half

    // -------- phase 1: hid = relu(x @ W1 + b1), 16 N-tiles over 8 waves --------
    for (int t = wave; t < 16; t += 8) {
        v8f acc = {};
        const int n = t * 16 + cB;
        for (int k0 = 0; k0 < F_IN; k0 += 32) {
            const float* ap = x + (size_t)(m0 + rA) * F_IN + k0 + kbA;
            v4f f0 = *(const v4f*)(ap);
            v4f f1 = *(const v4f*)(ap + 4);
            v4f f2 = *(const v4f*)(ap + 16);
            v4f f3 = *(const v4f*)(ap + 20);
            v16bf a;
#pragma unroll
            for (int e = 0; e < 4; ++e) {
                a[e]      = (__bf16)f0[e];
                a[4 + e]  = (__bf16)f1[e];
                a[8 + e]  = (__bf16)f2[e];
                a[12 + e] = (__bf16)f3[e];
            }
            v8bf b0 = *(const v8bf*)(w1t + (size_t)n * F_IN + k0 + kbB);
            v8bf b1v = *(const v8bf*)(w1t + (size_t)n * F_IN + k0 + kbB + 16);
            v16bf b = __builtin_shufflevector(b0, b1v, 0, 1, 2, 3, 4, 5, 6, 7, 8, 9, 10, 11, 12, 13, 14, 15);
            acc = __builtin_amdgcn_wmma_f32_16x16x32_bf16(
                false, a, false, b, (short)0, acc, false, false);
        }
        const float bias = b1[n];
#pragma unroll
        for (int r = 0; r < 8; ++r) {
            float v = acc[r] + bias;
            hid[mo + r][n] = (__bf16)(v > 0.0f ? v : 0.0f);
        }
    }
    __syncthreads();

    // -------- phase 2: h = hid @ W2 + b2, 4 N-tiles on waves 0..3 --------
    if (wave < 4) {
        v8f acc = {};
        const int n = wave * 16 + cB;
        for (int k0 = 0; k0 < F_HID; k0 += 32) {
            v8bf a0 = *(const v8bf*)&hid[rA][k0 + kbA];
            v8bf a1 = *(const v8bf*)&hid[rA][k0 + kbA + 16];
            v16bf a;
#pragma unroll
            for (int e = 0; e < 8; ++e) {
                a[e]     = a0[e];
                a[8 + e] = a1[e];
            }
            v8bf b0 = *(const v8bf*)(w2t + (size_t)n * F_HID + k0 + kbB);
            v8bf b1v = *(const v8bf*)(w2t + (size_t)n * F_HID + k0 + kbB + 16);
            v16bf b = __builtin_shufflevector(b0, b1v, 0, 1, 2, 3, 4, 5, 6, 7, 8, 9, 10, 11, 12, 13, 14, 15);
            acc = __builtin_amdgcn_wmma_f32_16x16x32_bf16(
                false, a, false, b, (short)0, acc, false, false);
        }
        const float bias = b2[n];
#pragma unroll
        for (int r = 0; r < 8; ++r)
            h[(size_t)(m0 + mo + r) * F_OUT + n] = acc[r] + bias;
    }
}

// ---------------------------------------------------------------- propagation
// znext = (1-a)*dinv[i]^2*z[i] + a*h[i]   (self-loop term folded into dense init)
__global__ __launch_bounds__(256) void prop_init(const float* __restrict__ z,
                                                 const float* __restrict__ h,
                                                 const float* __restrict__ dinv,
                                                 float* __restrict__ znext) {
    int t = blockIdx.x * 256 + threadIdx.x;              // N*64 = 6.4M
    if (t < N_NODES * F_OUT) {
        int i = t >> 6;
        float di = dinv[i];
        znext[t] = (1.0f - ALPHA) * di * di * z[t] + ALPHA * h[t];
    }
}

__global__ __launch_bounds__(256) void prop_scatter(const int* __restrict__ row,
                                                    const int* __restrict__ col,
                                                    const float* __restrict__ wn,
                                                    const float* __restrict__ z,
                                                    float* __restrict__ znext) {
    int t = blockIdx.x * 256 + threadIdx.x;              // E*64 = 204.8M < 2^31
    int e = t >> 6;
    int f = t & 63;
    if (e < N_EDGES) {
        int r = row[e];
        int c = col[e];
        unsafeAtomicAdd(&znext[(size_t)c * F_OUT + f], wn[e] * z[(size_t)r * F_OUT + f]);
    }
}

// ---------------------------------------------------------------- log_softmax (in place)
__global__ __launch_bounds__(256) void logsoftmax_kernel(float* __restrict__ z) {
    const int lane = threadIdx.x & 31;
    const int wave = threadIdx.x >> 5;
    const int rowi = blockIdx.x * 8 + wave;
    if (rowi >= N_NODES) return;
    float* p = z + (size_t)rowi * F_OUT;
    float v0 = p[lane];
    float v1 = p[lane + 32];
    float m = fmaxf(v0, v1);
#pragma unroll
    for (int o = 16; o > 0; o >>= 1) m = fmaxf(m, __shfl_xor(m, o, 32));
    float s = __expf(v0 - m) + __expf(v1 - m);
#pragma unroll
    for (int o = 16; o > 0; o >>= 1) s += __shfl_xor(s, o, 32);
    float l = m + __logf(s);
    p[lane]      = v0 - l;
    p[lane + 32] = v1 - l;
}

// ---------------------------------------------------------------- launch
extern "C" void kernel_launch(void* const* d_in, const int* in_sizes, int n_in,
                              void* d_out, int out_size, void* d_ws, size_t ws_size,
                              hipStream_t stream) {
    const float* x   = (const float*)d_in[0];
    const int*   ei  = (const int*)d_in[1];
    const float* W1  = (const float*)d_in[2];
    const float* b1  = (const float*)d_in[3];
    const float* W2  = (const float*)d_in[4];
    const float* b2  = (const float*)d_in[5];
    const int* row = ei;
    const int* col = ei + N_EDGES;

    // workspace carve-up (256B aligned)
    char* ws = (char*)d_ws;
    size_t off = 0;
    auto carve = [&](size_t bytes) {
        char* p = ws + off;
        off += (bytes + 255) & ~(size_t)255;
        return p;
    };
    float*  dinv  = (float*)carve((size_t)N_NODES * 4);
    float*  h     = (float*)carve((size_t)N_NODES * F_OUT * 4);
    float*  zA    = (float*)carve((size_t)N_NODES * F_OUT * 4);
    float*  wnorm = (float*)carve((size_t)N_EDGES * 4);
    __bf16* w1t   = (__bf16*)carve((size_t)F_HID * F_IN * 2);
    __bf16* w2t   = (__bf16*)carve((size_t)F_OUT * F_HID * 2);
    float*  zB    = (float*)d_out;                       // final step lands in d_out

    // weights -> transposed bf16
    prep_w1<<<(F_HID * F_IN + 255) / 256, 256, 0, stream>>>(W1, w1t);
    prep_w2<<<(F_OUT * F_HID + 255) / 256, 256, 0, stream>>>(W2, w2t);

    // gcn_norm
    deg_init<<<(N_NODES + 255) / 256, 256, 0, stream>>>(dinv);
    deg_scatter<<<(N_EDGES + 255) / 256, 256, 0, stream>>>(col, dinv);
    dinv_kernel<<<(N_NODES + 255) / 256, 256, 0, stream>>>(dinv);
    wnorm_kernel<<<(N_EDGES + 255) / 256, 256, 0, stream>>>(row, col, dinv, wnorm);

    // MLP (WMMA): 100000 rows / 16 = 6250 blocks, no remainder
    mlp_kernel<<<N_NODES / 16, 256, 0, stream>>>(x, w1t, b1, w2t, b2, h);

    // K_STEPS propagation rounds, ping-pong zA <-> zB(d_out)
    const float* zcur = h;
    for (int s = 0; s < K_STEPS; ++s) {
        float* znext = (s & 1) ? zB : zA;                // s=9 (last) -> zB = d_out
        prop_init<<<(N_NODES * F_OUT + 255) / 256, 256, 0, stream>>>(zcur, h, dinv, znext);
        prop_scatter<<<(int)(((size_t)N_EDGES * F_OUT + 255) / 256), 256, 0, stream>>>(
            row, col, wnorm, zcur, znext);
        zcur = znext;
    }

    // log_softmax in place on d_out
    logsoftmax_kernel<<<(N_NODES + 7) / 8, 256, 0, stream>>>((float*)d_out);
}
